// CNNQNetwork_89928025243724
// MI455X (gfx1250) — compile-verified, matrix operand order checked by reading, not run on hardware
//
#include <hip/hip_runtime.h>

// ---------------------------------------------------------------------------
// CNN Q-network, fused for MI455X (gfx1250, wave32, WMMA).
//   FC1 (122 GFLOP) dominates -> bf16 v_wmma_f32_16x16x32_bf16.
//   conv+FC1 fused per 64-row batch tile: conv (VALU, unrolled) writes directly
//   into CDNA5 A-fragment layout in LDS; WMMA consumes slabs immediately.
//   fw0 (3.7 MB bf16) is L2-resident -> B fragments gathered from global,
//   with global_prefetch warming the next K-slab.
//   h1 epilogue staged through LDS -> coalesced b128 stores.
// Workspace: fw0 bf16 (3.65 MB) + fw1 bf16 (128 KB) + h1 bf16 (32 MB) ~ 37 MB.
// ---------------------------------------------------------------------------

typedef __bf16 bf16_t;
typedef __attribute__((ext_vector_type(16))) __bf16 v16bf;
typedef __attribute__((ext_vector_type(8)))  float  v8f;

#define THREADS 256
#define M_TILE  64

static __device__ __forceinline__ v8f wmma_bf16(v16bf a, v16bf b, v8f c) {
  // (neg_a, A, neg_b, B, c_mod, C, reuse_a, reuse_b)
  return __builtin_amdgcn_wmma_f32_16x16x32_bf16(false, a, false, b, (short)0, c,
                                                 false, false);
}

// Gather this lane's 16 bf16 elements of a 16x32 (rows x K) WMMA operand
// fragment from a row-major matrix. rowptr = &M[row*stride + k0], where row
// already includes (lane & 15). Per ISA 7.12.2: lanes 0-15 hold K {0..7,16..23},
// lanes 16-31 hold K {8..15,24..31}, two bf16 per VGPR (dword = K pair).
// Dword pattern is contiguous 16B + 16B -> lowers to two global_load_b128.
static __device__ __forceinline__ v16bf frag_rowmajor(const bf16_t* rowptr, int lane) {
  const uint32_t* p = (const uint32_t*)rowptr;  // dword-aligned: k0 even
  const int o = (lane >> 4) << 2;               // +8 K for upper half-wave
  union { uint32_t u[8]; v16bf v; } f;
#pragma unroll
  for (int j = 0; j < 4; ++j) {
    f.u[j]     = p[o + j];       // K = o*2 + {0..7}
    f.u[4 + j] = p[o + 8 + j];   // K = 16 + o*2 + {0..7}
  }
  return f.v;
}

// ---------------------------------------------------------------------------
// One conv branch: VALU conv (ReLU) -> A-fragment LDS slabs -> WMMA into acc.
// acc[2][8]: this wave's 2 M-frags x 8 N-frags of the [64 x 512] FC1 tile.
// ---------------------------------------------------------------------------
template <int KH, int KW>
static __device__ __forceinline__ void branch_fc1(
    const float* __restrict__ w, const float* __restrict__ bias, int colbase,
    const bf16_t* __restrict__ fw0b, const float* xs, bf16_t* Ast,
    v8f acc[2][8], int tid, int lane, int wm, int wn)
{
  constexpr int OH = 5 - KH, OW = 5 - KW, OHW = OH * OW;
  constexpr int KCOLS = 64 * OHW;      // columns this branch contributes
  constexpr int NSLAB = KCOLS / 32;    // all branch sizes are multiples of 32

  for (int c0 = 0; c0 < NSLAB; c0 += 8) {
    const int ns = (NSLAB - c0 < 8) ? (NSLAB - c0) : 8;
    const int ccols = ns * 32;
    __syncthreads();  // previous slabs fully consumed; Ast reusable

    // ---- conv directly into WMMA A-fragment layout (bf16) in LDS ----
    for (int idx = tid; idx < M_TILE * ccols; idx += THREADS) {
      const int cc = idx % ccols;      // column within chunk (varies per lane)
      const int m  = idx / ccols;      // batch row within tile
      const int c  = c0 * 32 + cc;
      const int oc = c / OHW;
      const int p  = c % OHW;
      const int ph = p / OW, pw = p % OW;
      const float* wp = w + oc * (16 * KH * KW);      // OIHW
      const float* xp = xs + m * 256 + ph * 4 + pw;   // [16][4][4] per sample
      float a = bias[oc];
#pragma unroll
      for (int ci = 0; ci < 16; ++ci)
#pragma unroll
        for (int dh = 0; dh < KH; ++dh)
#pragma unroll
          for (int dw = 0; dw < KW; ++dw)
            a = fmaf(xp[ci * 16 + dh * 4 + dw], wp[(ci * KH + dh) * KW + dw], a);
      a = fmaxf(a, 0.0f);
      // inverse of the A-fragment K mapping: kin -> (element e, lane half)
      const int kin = cc & 31, slab = cc >> 5;
      const int e  = (kin & 7) | ((kin >> 1) & 8);
      const int lf = (m & 15) | (((kin >> 3) & 1) << 4);
      Ast[(slab * 4 + (m >> 4)) * 512 + lf * 16 + e] = (bf16_t)a;
    }
    __syncthreads();

    // ---- WMMA accumulate over the chunk's 32-K slabs ----
    for (int s = 0; s < ns; ++s) {
      const int kglob = colbase + (c0 + s) * 32;
      v16bf a0 = *(const v16bf*)(Ast + (s * 4 + 2 * wm + 0) * 512 + lane * 16);
      v16bf a1 = *(const v16bf*)(Ast + (s * 4 + 2 * wm + 1) * 512 + lane * 16);
#pragma unroll
      for (int j = 0; j < 8; ++j) {
        const int n = (wn * 8 + j) * 16 + (lane & 15);
        const bf16_t* bptr = fw0b + (size_t)n * 3648 + kglob;
        // warm the next 32-K slab of this fw0 row (global_prefetch_b8)
        __builtin_prefetch((const char*)bptr + 64, 0, 1);
        v16bf bf = frag_rowmajor(bptr, lane);
        acc[0][j] = wmma_bf16(a0, bf, acc[0][j]);
        acc[1][j] = wmma_bf16(a1, bf, acc[1][j]);
      }
    }
  }
}

// ---------------------------------------------------------------------------
// Kernel 1: fused 7-branch conv + FC1 (+bias+ReLU), h1 out in bf16.
// Block = 256 threads (8 waves), tile = 64 batch rows x all 512 FC1 outputs.
// ---------------------------------------------------------------------------
__global__ __launch_bounds__(THREADS) void k_conv_fc1(
    const float* __restrict__ x,
    const float* __restrict__ w0, const float* __restrict__ b0,
    const float* __restrict__ w1, const float* __restrict__ b1,
    const float* __restrict__ w2, const float* __restrict__ b2,
    const float* __restrict__ w3, const float* __restrict__ b3,
    const float* __restrict__ w4, const float* __restrict__ b4,
    const float* __restrict__ w5, const float* __restrict__ b5,
    const float* __restrict__ w6, const float* __restrict__ b6,
    const bf16_t* __restrict__ fw0b, const float* __restrict__ fb0,
    bf16_t* __restrict__ h1)
{
  __shared__ float  xs[M_TILE * 256];     // 64 KB: x tile (later: h1 stage)
  __shared__ bf16_t Ast[8 * 4 * 512];     // 32 KB: 8 K-slabs of A fragments

  const int tid  = threadIdx.x;
  const int lane = tid & 31;
  const int wave = tid >> 5;
  const int wm = wave & 1;                // M-frag pair: rows [32*wm, 32*wm+32)
  const int wn = wave >> 1;               // N range: cols [128*wn, 128*wn+128)
  const int m0 = blockIdx.x * M_TILE;

  {  // stage x tile, coalesced 128-bit loads
    const float4* g = (const float4*)(x + (size_t)m0 * 256);
    float4* s = (float4*)xs;
    for (int i = tid; i < M_TILE * 64; i += THREADS) s[i] = g[i];
  }

  v8f acc[2][8];
  const v8f vzero = {0, 0, 0, 0, 0, 0, 0, 0};
#pragma unroll
  for (int i = 0; i < 2; ++i)
#pragma unroll
    for (int j = 0; j < 8; ++j) acc[i][j] = vzero;

  // concat column offsets: 768,768,512,512,256,256,576
  branch_fc1<1, 2>(w0, b0,    0, fw0b, xs, Ast, acc, tid, lane, wm, wn);
  branch_fc1<2, 1>(w1, b1,  768, fw0b, xs, Ast, acc, tid, lane, wm, wn);
  branch_fc1<1, 3>(w2, b2, 1536, fw0b, xs, Ast, acc, tid, lane, wm, wn);
  branch_fc1<3, 1>(w3, b3, 2048, fw0b, xs, Ast, acc, tid, lane, wm, wn);
  branch_fc1<1, 4>(w4, b4, 2560, fw0b, xs, Ast, acc, tid, lane, wm, wn);
  branch_fc1<4, 1>(w5, b5, 2816, fw0b, xs, Ast, acc, tid, lane, wm, wn);
  branch_fc1<2, 2>(w6, b6, 3072, fw0b, xs, Ast, acc, tid, lane, wm, wn);

  // epilogue: h1 = relu(acc + fb0) -> bf16, staged in LDS (x tile is dead;
  // every wave passed the post-conv barrier of the last branch, and only Ast
  // is read afterwards, so reusing xs here is race-free).
  bf16_t* hst = (bf16_t*)xs;              // [64 x 512] bf16 = 64 KB
#pragma unroll
  for (int j = 0; j < 8; ++j) {
    const int n = (wn * 8 + j) * 16 + (lane & 15);
    const float bv = fb0[n];
#pragma unroll
    for (int mi = 0; mi < 2; ++mi)
#pragma unroll
      for (int v = 0; v < 8; ++v) {
        // C layout: N = lane&15, M = vgpr + 8*(lane>=16) within 16x16 frag
        const int m = (2 * wm + mi) * 16 + v + ((lane >> 4) << 3);
        hst[m * 512 + n] = (bf16_t)fmaxf(acc[mi][j][v] + bv, 0.0f);
      }
  }
  __syncthreads();
  {  // coalesced 128-bit stores of the [64 x 512] bf16 tile
    const uint4* s = (const uint4*)hst;
    uint4* g = (uint4*)(h1 + (size_t)m0 * 512);
#pragma unroll
    for (int i = 0; i < (M_TILE * 512 / 8) / THREADS; ++i)
      g[i * THREADS + tid] = s[i * THREADS + tid];
  }
}

// ---------------------------------------------------------------------------
// Kernel 2: FC2 (WMMA, K=512, +bias+ReLU) -> LDS h2 tile -> FC3 (128->4, VALU).
// Block tile: 128 rows x 128 cols (full FC2 width).
// ---------------------------------------------------------------------------
__global__ __launch_bounds__(THREADS) void k_fc2_fc3(
    const bf16_t* __restrict__ h1, const bf16_t* __restrict__ fw1b,
    const float* __restrict__ fb1, const float* __restrict__ fw2,
    const float* __restrict__ fb2, float* __restrict__ out)
{
  __shared__ bf16_t Afr[8 * 32 * 16];   // 8 KB: one K-slab of A fragments
  __shared__ float  h2s[128 * 132];     // 66 KB: padded h2 tile

  const int tid = threadIdx.x, lane = tid & 31, wave = tid >> 5;
  const int wm = wave & 3, wn = wave >> 2;
  const int m0 = blockIdx.x * 128;

  v8f acc[2][4];
  const v8f vzero = {0, 0, 0, 0, 0, 0, 0, 0};
#pragma unroll
  for (int i = 0; i < 2; ++i)
#pragma unroll
    for (int j = 0; j < 4; ++j) acc[i][j] = vzero;

  for (int s = 0; s < 16; ++s) {        // K = 512 -> 16 slabs of 32
    __syncthreads();
    {  // cooperative A staging: thread (f, l) gathers lane l of M-frag f
      const int f = tid >> 5, l = tid & 31;
      const int row = m0 + f * 16 + (l & 15);
      v16bf a = frag_rowmajor(h1 + (size_t)row * 512 + s * 32, l);
      *(v16bf*)(Afr + (f * 32 + l) * 16) = a;
    }
    __syncthreads();
    v16bf a0 = *(const v16bf*)(Afr + ((2 * wm + 0) * 32 + lane) * 16);
    v16bf a1 = *(const v16bf*)(Afr + ((2 * wm + 1) * 32 + lane) * 16);
#pragma unroll
    for (int j = 0; j < 4; ++j) {
      const int n = (wn * 4 + j) * 16 + (lane & 15);
      const bf16_t* bptr = fw1b + (size_t)n * 512 + s * 32;
      __builtin_prefetch((const char*)bptr + 64, 0, 1);
      v16bf bf = frag_rowmajor(bptr, lane);
      acc[0][j] = wmma_bf16(a0, bf, acc[0][j]);
      acc[1][j] = wmma_bf16(a1, bf, acc[1][j]);
    }
  }

  // h2 = relu(acc + fb1) -> LDS (fp32, padded rows)
#pragma unroll
  for (int j = 0; j < 4; ++j) {
    const int n = (wn * 4 + j) * 16 + (lane & 15);
    const float bv = fb1[n];
#pragma unroll
    for (int mi = 0; mi < 2; ++mi)
#pragma unroll
      for (int v = 0; v < 8; ++v) {
        const int ml = (2 * wm + mi) * 16 + v + ((lane >> 4) << 3);
        h2s[ml * 132 + n] = fmaxf(acc[mi][j][v] + bv, 0.0f);
      }
  }
  __syncthreads();

  // FC3: 128 rows x 4 outputs, fp32
  for (int t = tid; t < 128 * 4; t += THREADS) {
    const int ml = t >> 2, n = t & 3;
    const float* wr = fw2 + n * 128;
    float a = fb2[n];
#pragma unroll 8
    for (int k = 0; k < 128; ++k) a = fmaf(h2s[ml * 132 + k], wr[k], a);
    out[(size_t)(m0 + ml) * 4 + n] = a;
  }
}

// ---------------------------------------------------------------------------
// Kernel 0: fp32 -> bf16 weight conversion
// ---------------------------------------------------------------------------
__global__ void k_cvt(const float* __restrict__ s, bf16_t* __restrict__ d, int n) {
  int i = blockIdx.x * 256 + threadIdx.x;
  if (i < n) d[i] = (bf16_t)s[i];
}

extern "C" void kernel_launch(void* const* d_in, const int* in_sizes, int n_in,
                              void* d_out, int out_size, void* d_ws, size_t ws_size,
                              hipStream_t stream) {
  (void)n_in; (void)out_size; (void)ws_size;
  const float* x = (const float*)d_in[0];
  const float* w[7]; const float* b[7];
  for (int i = 0; i < 7; ++i) {
    w[i] = (const float*)d_in[1 + 2 * i];
    b[i] = (const float*)d_in[2 + 2 * i];
  }
  const float* fw0 = (const float*)d_in[15];
  const float* fb0 = (const float*)d_in[16];
  const float* fw1 = (const float*)d_in[17];
  const float* fb1 = (const float*)d_in[18];
  const float* fw2 = (const float*)d_in[19];
  const float* fb2 = (const float*)d_in[20];
  float* out = (float*)d_out;
  const int B = in_sizes[0] / 256;  // 32768

  // workspace layout (bytes): fw0b | fw1b | h1   (~37.4 MB total)
  char* ws = (char*)d_ws;
  bf16_t* fw0b = (bf16_t*)ws;
  bf16_t* fw1b = (bf16_t*)(ws + (size_t)512 * 3648 * 2);
  bf16_t* h1   = (bf16_t*)(ws + (size_t)512 * 3648 * 2 + (size_t)128 * 512 * 2);

  const int n0 = 512 * 3648, n1 = 128 * 512;
  k_cvt<<<(n0 + 255) / 256, 256, 0, stream>>>(fw0, fw0b, n0);
  k_cvt<<<(n1 + 255) / 256, 256, 0, stream>>>(fw1, fw1b, n1);

  k_conv_fc1<<<B / M_TILE, THREADS, 0, stream>>>(
      x, w[0], b[0], w[1], b[1], w[2], b[2], w[3], b[3], w[4], b[4],
      w[5], b[5], w[6], b[6], fw0b, fb0, h1);

  k_fc2_fc3<<<B / 128, THREADS, 0, stream>>>(h1, fw1b, fb1, fw2, fb2, out);
}
